// PairwiseSimilarityLossParallel_31748398252480
// MI455X (gfx1250) — compile-verified, matrix operand order checked by reading, not run on hardware
//
#include <hip/hip_runtime.h>
#include <hip/hip_bf16.h>

// ---------------------------------------------------------------------------
// PairwiseSimilarityLoss on MI455X (gfx1250)
//   S_full = cos-sim(embeddings)  [4096x4096], K=1024
//   S_m    = cos-sim(adapted[:, :m]) for m in {64,128,256,512}
//   loss   = masked (j>=i) sum of |S_m - S_full|
//
// f16 WMMA 16x16x32 / f32 accumulate. Embeddings pre-normalized so the raw
// accumulator IS S_full. Adapted accumulated once over K=512 with prefix
// checkpoints at K=64/128/256/512 rescaled by precomputed inverse norms.
// Block = 8 waves = 128x64 macro-tile of the upper triangle; per K-step the
// block DMAs A(128x32)+B(64x32) f16 into LDS with global_load_async_to_lds_b128
// (double buffered, s_wait_asynccnt). Each wave register-tiles 32x32:
// 2 A-frags x 2 B-frags -> 4 WMMAs per step (1 KB LDS per WMMA,
// 44 FLOP/byte from L2). Sub-diagonal work is handled purely by the
// per-element mask (no EXEC-masked accumulator regions -> no register copies).
// ---------------------------------------------------------------------------

typedef _Float16 v4h  __attribute__((ext_vector_type(4)));
typedef _Float16 v8h  __attribute__((ext_vector_type(8)));
typedef _Float16 v16h __attribute__((ext_vector_type(16)));
typedef float    v8f  __attribute__((ext_vector_type(8)));

#define N_ROWS   4096
#define D_FULL   1024
#define D_ADP    512
#define NRI      32            // 4096/128 row macro-blocks
#define NBLK     1056          // sum over ri of (64 - 2*ri) col blocks
#define LDSS     40            // staged LDS row stride in halfs (32 data + 8 pad = 80B)

// ws layout (bytes)
#define WS_EH    0             // 4096*1024 f16 = 8 MB (normalized embeddings)
#define WS_AH    8388608       // 4096*512  f16 = 4 MB (adapted, first 512 dims)
#define WS_INVA  12582912      // 4*4096 f32 (1/||a[:m]|| per m, per row)
#define WS_PART  12648448      // 4 f32 partial sums

// ---------------------------------------------------------------------------
// Prep: row norms, f16 conversion, prefix inverse norms. One block per row.
// ---------------------------------------------------------------------------
__global__ void __launch_bounds__(256)
prep_kernel(const float* __restrict__ emb, const float* __restrict__ adp,
            _Float16* __restrict__ Eh, _Float16* __restrict__ Ah,
            float* __restrict__ invA) {
  const int row = blockIdx.x;
  const int tid = threadIdx.x;
  __shared__ float red[256];
  __shared__ float seg[4];
  __shared__ float s_invE;

  const float4 ev = ((const float4*)(emb + (size_t)row * D_FULL))[tid];
  red[tid] = ev.x * ev.x + ev.y * ev.y + ev.z * ev.z + ev.w * ev.w;
  if (tid < 4) seg[tid] = 0.0f;
  __syncthreads();
  for (int s = 128; s > 0; s >>= 1) {
    if (tid < s) red[tid] += red[tid + s];
    __syncthreads();
  }
  if (tid == 0) s_invE = 1.0f / fmaxf(sqrtf(red[0]), 1e-8f);
  __syncthreads();
  const float invE = s_invE;

  {
    v4h w;
    w.x = (_Float16)(ev.x * invE);
    w.y = (_Float16)(ev.y * invE);
    w.z = (_Float16)(ev.z * invE);
    w.w = (_Float16)(ev.w * invE);
    ((v4h*)(Eh + (size_t)row * D_FULL))[tid] = w;
  }

  if (tid < 128) {
    const float4 av = ((const float4*)(adp + (size_t)row * D_FULL))[tid];
    const float la = av.x * av.x + av.y * av.y + av.z * av.z + av.w * av.w;
    const int k0 = tid * 4;
    const int sidx = (k0 < 64) ? 0 : (k0 < 128) ? 1 : (k0 < 256) ? 2 : 3;
    atomicAdd(&seg[sidx], la);
    v4h w;
    w.x = (_Float16)av.x; w.y = (_Float16)av.y;
    w.z = (_Float16)av.z; w.w = (_Float16)av.w;
    ((v4h*)(Ah + (size_t)row * D_ADP))[tid] = w;
  }
  __syncthreads();
  if (tid < 4) {
    float c = seg[0];
    if (tid >= 1) c += seg[1];
    if (tid >= 2) c += seg[2];
    if (tid >= 3) c += seg[3];
    invA[tid * N_ROWS + row] = 1.0f / fmaxf(sqrtf(c), 1e-8f);
  }
}

// ---------------------------------------------------------------------------
// blocks per row-band prefix: off(r) = sum_{k<r} (64-2k) = r*(65-r)
__device__ __forceinline__ int rb_off(int r) { return r * (65 - r); }

__device__ __forceinline__ v16h lds_frag(const _Float16* p) {
  // f16 A/B fragment: lane reads K-chunks [half*8,+8) and [16+half*8,+8);
  // p is pre-offset by half*8, so two 16-byte reads at +0 and +16 elems.
  v8h lo = *(const v8h*)p;
  v8h hi = *(const v8h*)(p + 16);
  return __builtin_shufflevector(lo, hi, 0, 1, 2, 3, 4, 5, 6, 7,
                                 8, 9, 10, 11, 12, 13, 14, 15);
}

__global__ void __launch_bounds__(256)
tile_kernel(const _Float16* __restrict__ Eh, const _Float16* __restrict__ Ah,
            const float* __restrict__ invA, float* __restrict__ partials) {
  __shared__ __attribute__((aligned(16))) _Float16 smA[2][128 * LDSS];
  __shared__ __attribute__((aligned(16))) _Float16 smB[2][64 * LDSS];
  __shared__ float red[4][256];

  const int tid  = threadIdx.x;
  const int lane = tid & 31;
  const int wave = tid >> 5;
  const int sr   = wave & 3;        // 32-row band within macro tile (4 bands)
  const int sc   = wave >> 2;       // 32-col band within macro tile (2 bands)
  const int mrow = lane & 15, half = lane >> 4;

  // map blockIdx -> (ri, cj): 128-row band ri, 64-col block cj >= 2*ri
  const int t = blockIdx.x;
  int ri = (int)((65.0f - sqrtf(4225.0f - 4.0f * (float)t)) * 0.5f);
  ri = ri < 0 ? 0 : (ri > NRI - 1 ? NRI - 1 : ri);
  while (rb_off(ri + 1) <= t) ++ri;
  while (rb_off(ri) > t) --ri;
  const int cj = 2 * ri + (t - rb_off(ri));
  const int i0 = ri * 128, j0 = cj * 64;

  // async DMA of one K=32 slab: A rows i0..i0+127, B rows j0..j0+63.
  // 768 16B chunks / 256 threads = 3 async loads per thread per step.
  const int strow = tid >> 2, stseg = (tid & 3) * 8;
  auto stage = [&](const _Float16* bi, const _Float16* bj, size_t ld,
                   int kk, int buf) {
    const _Float16* ga0 = bi + (size_t)strow * ld + kk + stseg;
    const _Float16* ga1 = bi + (size_t)(strow + 64) * ld + kk + stseg;
    const _Float16* gb  = bj + (size_t)strow * ld + kk + stseg;
    const unsigned la0 = (unsigned)(size_t)&smA[buf][strow * LDSS + stseg];
    const unsigned la1 = (unsigned)(size_t)&smA[buf][(strow + 64) * LDSS + stseg];
    const unsigned lb  = (unsigned)(size_t)&smB[buf][strow * LDSS + stseg];
    asm volatile("global_load_async_to_lds_b128 %0, %1, off"
                 :: "v"(la0), "v"(ga0) : "memory");
    asm volatile("global_load_async_to_lds_b128 %0, %1, off"
                 :: "v"(la1), "v"(ga1) : "memory");
    asm volatile("global_load_async_to_lds_b128 %0, %1, off"
                 :: "v"(lb), "v"(gb) : "memory");
  };

  const int arow0 = (sr * 32 + mrow) * LDSS + half * 8;
  const int arow1 = (sr * 32 + 16 + mrow) * LDSS + half * 8;
  const int brow0 = (sc * 32 + mrow) * LDSS + half * 8;
  const int brow1 = (sc * 32 + 16 + mrow) * LDSS + half * 8;
  auto mac4 = [&](int buf, v8f& c00, v8f& c01, v8f& c10, v8f& c11) {
    v16h a0 = lds_frag(&smA[buf][arow0]);
    v16h a1 = lds_frag(&smA[buf][arow1]);
    v16h b0 = lds_frag(&smB[buf][brow0]);
    v16h b1 = lds_frag(&smB[buf][brow1]);
    c00 = __builtin_amdgcn_wmma_f32_16x16x32_f16(false, a0, false, b0,
                                                 (short)0, c00, false, false);
    c01 = __builtin_amdgcn_wmma_f32_16x16x32_f16(false, a0, false, b1,
                                                 (short)0, c01, false, false);
    c10 = __builtin_amdgcn_wmma_f32_16x16x32_f16(false, a1, false, b0,
                                                 (short)0, c10, false, false);
    c11 = __builtin_amdgcn_wmma_f32_16x16x32_f16(false, a1, false, b1,
                                                 (short)0, c11, false, false);
  };

  // ---- phase 1: S_full, K = 1024 (32 steps, double buffered) ----
  const _Float16* Ei = Eh + (size_t)i0 * D_FULL;
  const _Float16* Ej = Eh + (size_t)j0 * D_FULL;
  v8f f00 = {}, f01 = {}, f10 = {}, f11 = {};
  stage(Ei, Ej, D_FULL, 0, 0);
  for (int s = 0; s < 32; ++s) {
    asm volatile("s_wait_asynccnt 0x0" ::: "memory");
    __syncthreads();                       // slab ready + all prior reads done
    if (s + 1 < 32) stage(Ei, Ej, D_FULL, (s + 1) * 32, (s + 1) & 1);
    mac4(s & 1, f00, f01, f10, f11);
  }

  // ---- phase 2: adapted, K = 512, prefix checkpoints m = 64/128/256/512 ----
  const int jc0 = j0 + sc * 32 + mrow;     // C/D layout: lane&15 = column N
  const int jc1 = jc0 + 16;
  const int ib0 = i0 + sr * 32 + half * 8; // VGPR r holds row M = half*8 + r
  const int ib1 = ib0 + 16;
  v8f a00 = {}, a01 = {}, a10 = {}, a11 = {};
  auto ckpt = [&](int c) -> float {
    const float* ia0  = invA + c * N_ROWS + ib0;
    const float* ia1  = invA + c * N_ROWS + ib1;
    const float invj0 = invA[c * N_ROWS + jc0];
    const float invj1 = invA[c * N_ROWS + jc1];
    float s = 0.0f;
#pragma unroll
    for (int r = 0; r < 8; ++r) {
      const float w0 = ia0[r], w1 = ia1[r];
      s += (jc0 >= ib0 + r) ? fabsf(a00[r] * (w0 * invj0) - f00[r]) : 0.0f;
      s += (jc1 >= ib0 + r) ? fabsf(a01[r] * (w0 * invj1) - f01[r]) : 0.0f;
      s += (jc0 >= ib1 + r) ? fabsf(a10[r] * (w1 * invj0) - f10[r]) : 0.0f;
      s += (jc1 >= ib1 + r) ? fabsf(a11[r] * (w1 * invj1) - f11[r]) : 0.0f;
    }
    return s;
  };

  const _Float16* Ai = Ah + (size_t)i0 * D_ADP;
  const _Float16* Aj = Ah + (size_t)j0 * D_ADP;
  float ms0 = 0.0f, ms1 = 0.0f, ms2 = 0.0f, ms3 = 0.0f;
  __syncthreads();
  stage(Ai, Aj, D_ADP, 0, 0);
  for (int s = 0; s < 16; ++s) {
    asm volatile("s_wait_asynccnt 0x0" ::: "memory");
    __syncthreads();
    if (s + 1 < 16) stage(Ai, Aj, D_ADP, (s + 1) * 32, (s + 1) & 1);
    mac4(s & 1, a00, a01, a10, a11);
    if (s == 1)       ms0 = ckpt(0);       // m = 64
    else if (s == 3)  ms1 = ckpt(1);       // m = 128
    else if (s == 7)  ms2 = ckpt(2);       // m = 256
    else if (s == 15) ms3 = ckpt(3);       // m = 512
  }

  // ---- block reduction, 4 atomics per block ----
  __syncthreads();
  red[0][tid] = ms0; red[1][tid] = ms1;
  red[2][tid] = ms2; red[3][tid] = ms3;
  __syncthreads();
  for (int s = 128; s > 0; s >>= 1) {
    if (tid < s) {
      red[0][tid] += red[0][tid + s];
      red[1][tid] += red[1][tid + s];
      red[2][tid] += red[2][tid + s];
      red[3][tid] += red[3][tid + s];
    }
    __syncthreads();
  }
  if (tid < 4) atomicAdd(&partials[tid], red[tid][0]);
}

// ---------------------------------------------------------------------------
__global__ void finalize_kernel(const float* __restrict__ partials,
                                float* __restrict__ out) {
  if (threadIdx.x == 0 && blockIdx.x == 0) {
    const float np = 8390656.0f;                 // 4096*4097/2
    const float p0 = partials[0], p1 = partials[1];
    const float p2 = partials[2], p3 = partials[3];
    out[0] = (p0 + p1 + p2 + p3) / (4.0f * np);  // loss / counter
    out[1] = p0 / np;
    out[2] = p1 / np;
    out[3] = p2 / np;
    out[4] = p3 / np;
  }
}

// ---------------------------------------------------------------------------
extern "C" void kernel_launch(void* const* d_in, const int* in_sizes, int n_in,
                              void* d_out, int out_size, void* d_ws, size_t ws_size,
                              hipStream_t stream) {
  const float* emb = (const float*)d_in[0];   // [4096,1024] f32
  const float* adp = (const float*)d_in[1];   // [4096,1024] f32
  // d_in[2] = m_list (fixed {64,128,256,512}; baked into the kernel)
  float* out = (float*)d_out;                 // 5 floats

  char* ws = (char*)d_ws;
  _Float16* Eh   = (_Float16*)(ws + WS_EH);
  _Float16* Ah   = (_Float16*)(ws + WS_AH);
  float*    invA = (float*)(ws + WS_INVA);
  float*    part = (float*)(ws + WS_PART);

  hipMemsetAsync(part, 0, 4 * sizeof(float), stream);
  prep_kernel<<<N_ROWS, 256, 0, stream>>>(emb, adp, Eh, Ah, invA);
  tile_kernel<<<NBLK, 256, 0, stream>>>(Eh, Ah, invA, part);
  finalize_kernel<<<1, 1, 0, stream>>>(part, out);
}
